// DeformConv2d_38946763440917
// MI455X (gfx1250) — compile-verified, matrix operand order checked by reading, not run on hardware
//
#include <hip/hip_runtime.h>
#include <hip/hip_bf16.h>

// ---------------------------------------------------------------------------
// Deformable Conv2d for MI455X (gfx1250), wave32, WMMA bf16 path.
// B=8, C=256, H=W=64, OUT=256, K=3 (K2=9), stride=1, pad=1, dil=1, OG=1.
// ---------------------------------------------------------------------------

#define B_   8
#define C_   256
#define HW_  4096          // 64*64
#define OUT_ 256
#define K2_  9
#define KDIM 2304          // C_*K2_
#define NCHUNK 4           // channel chunks for param conv

typedef __attribute__((ext_vector_type(16))) __bf16        v16bf;
typedef __attribute__((ext_vector_type(8)))  float         v8f;
typedef __attribute__((ext_vector_type(4)))  unsigned int  v4u;

union Frag { v16bf v; v4u q[2]; };

static __device__ __forceinline__ unsigned short f32_to_bf16(float f) {
    unsigned int u = __float_as_uint(f);
    unsigned int lsb = (u >> 16) & 1u;
    u += 0x7fffu + lsb;                 // round-to-nearest-even
    return (unsigned short)(u >> 16);
}

// ---------------------------------------------------------------------------
// Kernel 1: param-generator 3x3 conv, partial over channel chunk of 64.
// pp layout: [b][chunk][27][4096] f32
// ---------------------------------------------------------------------------
__global__ __launch_bounds__(256) void pg_conv(const float* __restrict__ x,
                                               const float* __restrict__ pgw,
                                               float* __restrict__ pp) {
    int id    = blockIdx.x * 256 + threadIdx.x;   // B_*NCHUNK*HW_ threads
    int p     = id & 4095;
    int chunk = (id >> 12) & 3;
    int b     = id >> 14;
    int h = p >> 6, w = p & 63;

    float acc[27];
#pragma unroll
    for (int i = 0; i < 27; ++i) acc[i] = 0.f;

    const float* xb = x + ((size_t)b * C_ + chunk * 64) * HW_;
    for (int c = 0; c < 64; ++c) {
        float xv[9];
        const float* xc = xb + c * HW_;
#pragma unroll
        for (int ti = 0; ti < 3; ++ti)
#pragma unroll
            for (int tj = 0; tj < 3; ++tj) {
                int yy = h + ti - 1, xx = w + tj - 1;
                bool ok = (yy >= 0) & (yy < 64) & (xx >= 0) & (xx < 64);
                xv[ti * 3 + tj] = ok ? xc[yy * 64 + xx] : 0.f;
            }
        int cg = chunk * 64 + c;
        const float* wp = pgw + (size_t)cg * 9;   // pgw[oc][c][3][3]
#pragma unroll
        for (int oc = 0; oc < 27; ++oc) {
            const float* wo = wp + (size_t)oc * KDIM;   // oc*C_*9
            float a = acc[oc];
#pragma unroll
            for (int t = 0; t < 9; ++t) a += xv[t] * wo[t];
            acc[oc] = a;
        }
    }
#pragma unroll
    for (int oc = 0; oc < 27; ++oc)
        pp[((((size_t)(b * NCHUNK + chunk)) * 27 + oc) << 12) + p] = acc[oc];
}

// ---------------------------------------------------------------------------
// Kernel 2: reduce chunks + bias, sigmoid mask, bilinear corner weights/indices
// with mask & validity folded into weights.  cW/cI layout: [b][j][4096]
// torchvision interleave: dy_j = ch 2j, dx_j = ch 2j+1, mask_j = ch 18+j
// ---------------------------------------------------------------------------
__global__ __launch_bounds__(256) void mk_corners(const float* __restrict__ pp,
                                                  const float* __restrict__ pgb,
                                                  float4* __restrict__ cW,
                                                  int4* __restrict__ cI) {
    int id = blockIdx.x * 256 + threadIdx.x;     // B_*K2_*HW_ threads
    int p  = id & 4095;
    int j  = (id >> 12) % K2_;
    int b  = id / (K2_ * HW_);

    float dy = pgb[2 * j], dx = pgb[2 * j + 1], m = pgb[18 + j];
#pragma unroll
    for (int ch = 0; ch < NCHUNK; ++ch) {
        const float* base = pp + (((size_t)(b * NCHUNK + ch) * 27) << 12) + p;
        dy += base[(size_t)(2 * j) << 12];
        dx += base[(size_t)(2 * j + 1) << 12];
        m  += base[(size_t)(18 + j) << 12];
    }
    float mask = 1.f / (1.f + __expf(-m));

    int h = p >> 6, w = p & 63;
    float py = dy + (float)(h - 1 + j / 3);
    float px = dx + (float)(w - 1 + j % 3);
    float y0f = floorf(py), x0f = floorf(px);
    float fy = py - y0f, fx = px - x0f;
    int y0 = (int)y0f, x0 = (int)x0f;

    float cw[4];
    int   ci[4];
    const float wy[2] = {1.f - fy, fy};
    const float wx[2] = {1.f - fx, fx};
#pragma unroll
    for (int oy = 0; oy < 2; ++oy)
#pragma unroll
        for (int ox = 0; ox < 2; ++ox) {
            int yy = y0 + oy, xx = x0 + ox;
            bool valid = (yy >= 0) & (yy < 64) & (xx >= 0) & (xx < 64);
            int yi = min(max(yy, 0), 63);
            int xi = min(max(xx, 0), 63);
            float wgt = wy[oy] * wx[ox] * mask;
            cw[oy * 2 + ox] = valid ? wgt : 0.f;
            ci[oy * 2 + ox] = yi * 64 + xi;
        }
    size_t g = ((size_t)b * K2_ + j) * HW_ + p;
    cW[g] = make_float4(cw[0], cw[1], cw[2], cw[3]);
    cI[g] = make_int4(ci[0], ci[1], ci[2], ci[3]);
}

// ---------------------------------------------------------------------------
// Kernel 3: main conv weight f32 -> bf16, row-major [OUT][KDIM], K = c*9 + j
// ---------------------------------------------------------------------------
__global__ __launch_bounds__(256) void cvt_w(const float* __restrict__ w,
                                             unsigned short* __restrict__ wb) {
    int i = blockIdx.x * 256 + threadIdx.x;
    if (i < OUT_ * KDIM) wb[i] = f32_to_bf16(w[i]);
}

// ---------------------------------------------------------------------------
// Kernel 4: fused bilinear-gather + GEMM.
//   one block = (batch b, 16-pixel tile). 8 waves; wave w owns M rows
//   [16w,16w+16) and [16w+128,16w+144). 72 K-steps of 32 over KDIM=2304.
//   S-tile (32 K x 16 pixels, bf16) built cooperatively in LDS per step.
// ---------------------------------------------------------------------------
__global__ __launch_bounds__(256) void deform_gemm(const float* __restrict__ x,
                                                   const unsigned short* __restrict__ wB,
                                                   const float* __restrict__ bias,
                                                   const float4* __restrict__ cW,
                                                   const int4* __restrict__ cI,
                                                   float* __restrict__ out) {
    int blk   = blockIdx.x;
    int b     = blk >> 8;        // 256 n-tiles per batch
    int ntile = blk & 255;
    int p0    = ntile * 16;

    __shared__ __align__(16) float4        sCW[K2_ * 16];
    __shared__ __align__(16) int4          sCI[K2_ * 16];
    __shared__ __align__(16) unsigned short sS[16][40];   // [n][k], padded

    int tid = threadIdx.x;
    if (tid < K2_ * 16) {
        int j = tid >> 4, n = tid & 15;
        size_t g = ((size_t)b * K2_ + j) * HW_ + p0 + n;
        sCW[tid] = cW[g];
        sCI[tid] = cI[g];
    }
    __syncthreads();

    int lane = tid & 31, wave = tid >> 5;
    int half = lane >> 4, l16 = lane & 15;
    int m0 = wave * 16, m1 = m0 + 128;

    v8f acc0 = {}, acc1 = {};
    const float* xb = x + (size_t)b * C_ * HW_;

    // gather assignment: this thread builds S[kk_a][n_a] and S[kk_a+16][n_a]
    int n_a  = tid & 15;
    int kk_a = tid >> 4;     // 0..15

    for (int ks = 0; ks < KDIM / 32; ++ks) {
        int K0 = ks * 32;
        // ---- build S tile (bilinear gather, mask folded into corner weights)
#pragma unroll
        for (int e = 0; e < 2; ++e) {
            int kk = kk_a + e * 16;
            int Ki = K0 + kk;
            int c  = Ki / 9, j = Ki % 9;
            float4 w4 = sCW[j * 16 + n_a];
            int4   i4 = sCI[j * 16 + n_a];
            const float* xc = xb + (size_t)c * HW_;
            float s = w4.x * xc[i4.x] + w4.y * xc[i4.y] +
                      w4.z * xc[i4.z] + w4.w * xc[i4.w];
            sS[n_a][kk] = f32_to_bf16(s);
        }
        __syncthreads();

        // ---- A frags from global bf16 weights (16x32, ISA 16-bit A layout)
        Frag A0, A1, Bf;
        const v4u* pb = (const v4u*)&sS[l16][half * 8];
        Bf.q[0] = pb[0];
        Bf.q[1] = pb[2];                 // +16 halfs
        const v4u* pa0 = (const v4u*)(wB + (size_t)(m0 + l16) * KDIM + K0 + half * 8);
        A0.q[0] = pa0[0];
        A0.q[1] = pa0[2];
        const v4u* pa1 = (const v4u*)(wB + (size_t)(m1 + l16) * KDIM + K0 + half * 8);
        A1.q[0] = pa1[0];
        A1.q[1] = pa1[2];

        acc0 = __builtin_amdgcn_wmma_f32_16x16x32_bf16(false, A0.v, false, Bf.v,
                                                       (short)0, acc0, false, false);
        acc1 = __builtin_amdgcn_wmma_f32_16x16x32_bf16(false, A1.v, false, Bf.v,
                                                       (short)0, acc1, false, false);
        __syncthreads();
    }

    // ---- store per ISA 16x16 f32 C layout: vgpr r -> M = r + 8*half, N = l16
    int col = p0 + l16;
#pragma unroll
    for (int r = 0; r < 8; ++r) {
        int rl = r + half * 8;
        int o0 = m0 + rl, o1 = m1 + rl;
        out[((size_t)b * OUT_ + o0) * HW_ + col] = acc0[r] + bias[o0];
        out[((size_t)b * OUT_ + o1) * HW_ + col] = acc1[r] + bias[o1];
    }
}

// ---------------------------------------------------------------------------
// launch
// ---------------------------------------------------------------------------
extern "C" void kernel_launch(void* const* d_in, const int* in_sizes, int n_in,
                              void* d_out, int out_size, void* d_ws, size_t ws_size,
                              hipStream_t stream) {
    const float* x    = (const float*)d_in[0];   // [8,256,64,64]
    const float* wgt  = (const float*)d_in[1];   // [256,256,3,3]
    const float* bias = (const float*)d_in[2];   // [256]
    const float* pg_w = (const float*)d_in[3];   // [27,256,3,3]
    const float* pg_b = (const float*)d_in[4];   // [27]
    float* out = (float*)d_out;

    char* ws = (char*)d_ws;
    size_t off_pp = 0;
    size_t sz_pp  = (size_t)B_ * NCHUNK * 27 * HW_ * sizeof(float);   // 14.2 MB
    size_t off_cw = off_pp + sz_pp;
    size_t sz_cw  = (size_t)B_ * K2_ * HW_ * sizeof(float4);          // 4.7 MB
    size_t off_ci = off_cw + sz_cw;
    size_t sz_ci  = (size_t)B_ * K2_ * HW_ * sizeof(int4);            // 4.7 MB
    size_t off_wb = off_ci + sz_ci;                                   // 1.2 MB

    float*          pp = (float*)(ws + off_pp);
    float4*         cw = (float4*)(ws + off_cw);
    int4*           ci = (int4*)(ws + off_ci);
    unsigned short* wb = (unsigned short*)(ws + off_wb);

    // 1) param-generator conv (chunked partials)
    pg_conv<<<(B_ * NCHUNK * HW_) / 256, 256, 0, stream>>>(x, pg_w, pp);
    // 2) offsets/mask -> bilinear corner descriptors
    mk_corners<<<(B_ * K2_ * HW_) / 256, 256, 0, stream>>>(pp, pg_b, cw, ci);
    // 3) weights -> bf16
    cvt_w<<<(OUT_ * KDIM + 255) / 256, 256, 0, stream>>>(wgt, wb);
    // 4) fused gather + WMMA GEMM
    deform_gemm<<<B_ * (HW_ / 16), 256, 0, stream>>>(x, wb, bias, cw, ci, out);
}
